// FrameGNN_73065983639597
// MI455X (gfx1250) — matrix operand order repaired and checked by reading, not compiled, not run on hardware
//
#include <hip/hip_runtime.h>

// ---------------------------------------------------------------------------
// FrameGNN forward for MI455X (gfx1250, wave32).
// Per wave: 16 batch elements. Scalar pre/mid/post phases on lanes 0-15,
// all six MLP GEMMs via v_wmma_f32_16x16x32_f16 (batch tile = M dimension).
// The mb_* MLP in the reference is dead code and is skipped.
// ---------------------------------------------------------------------------

typedef __attribute__((ext_vector_type(16))) _Float16 v16h;
typedef __attribute__((ext_vector_type(8)))  _Float16 v8h;
typedef __attribute__((ext_vector_type(8)))  float    v8f;

// ----- LDS layout (bytes) --------------------------------------------------
constexpr int XSTR   = 104;           // f16 per activation row (96 used + pad)
constexpr int XROWB  = XSTR * 2;      // 208 B  (52 dwords -> conflict-free)
constexpr int XWAVEB = 16 * XROWB;    // 3328 B per-wave activation tile

constexpr int OFF_W1 = 0;                    // up_m_w1  64 x 96 f16
constexpr int OFF_W2 = OFF_W1 + 64 * 96 * 2; // up_m_w2  64 x 64
constexpr int OFF_W3 = OFF_W2 + 64 * 64 * 2; // up_m_w3  80(pad) x 64
constexpr int OFF_F1 = OFF_W3 + 80 * 64 * 2; // fr_w1    64 x 96
constexpr int OFF_F2 = OFF_F1 + 64 * 96 * 2; // fr_w2    64 x 64
constexpr int OFF_F3 = OFF_F2 + 64 * 64 * 2; // fr_w3    16(pad) x 64
constexpr int OFF_B  = OFF_F3 + 16 * 64 * 2; // biases f32 (padded)
constexpr int BO_B1 = 0, BO_B2 = 64, BO_B3 = 128;     // b3 padded to 80
constexpr int BO_FB1 = 208, BO_FB2 = 272, BO_FB3 = 336; // fb3 padded to 16
constexpr int NBIAS = 352;
constexpr int OFF_SW = OFF_B + NBIAS * 4;    // small f32 weights
// small-weight sub-offsets (in floats)
constexpr int SW_UPZ1   = 0;    // 8x4  = 32
constexpr int SW_UPH1W  = 32;   // 8x13 = 104
constexpr int SW_UPH1B  = 136;  // 8
constexpr int SW_UPG1W1 = 144;  // 8x24 = 192
constexpr int SW_UPG1B1 = 336;  // 8
constexpr int SW_UPG1W2 = 344;  // 8
constexpr int SW_UPG1B2 = 352;  // 1
constexpr int SW_UPZ2   = 353;  // 8x25 = 200
constexpr int SW_DNG1W1 = 553;  // 8x16 = 128
constexpr int SW_DNG1B1 = 681;  // 8
constexpr int SW_DNG1W2 = 689;  // 8
constexpr int SW_DNG1B2 = 697;  // 1
constexpr int SW_DNZ2   = 698;  // 8x17 = 136
constexpr int NSW = 834;
constexpr int OFF_X = ((OFF_SW + NSW * 4) + 15) & ~15;
constexpr int SMEM_TOTAL = OFF_X + 8 * XWAVEB;   // ~84.6 KB (<< 320 KB WGP LDS)

// ----- block-cooperative staging helpers -----------------------------------
__device__ __forceinline__ void cpw(_Float16* dst, const float* src,
                                    int rows, int cols, int rpad, int cpad,
                                    int tid, int nthr) {
  const int tot = rpad * cpad;
  for (int i = tid; i < tot; i += nthr) {
    const int r = i / cpad, c = i - r * cpad;
    const float v = (r < rows && c < cols) ? src[r * cols + c] : 0.0f;
    dst[i] = (_Float16)v;
  }
}
__device__ __forceinline__ void cpb(float* dst, const float* src, int n, int npad,
                                    int tid, int nthr) {
  for (int i = tid; i < npad; i += nthr) dst[i] = (i < n) ? src[i] : 0.0f;
}
__device__ __forceinline__ void cpf(float* dst, const float* src, int n,
                                    int tid, int nthr) {
  for (int i = tid; i < n; i += nthr) dst[i] = src[i];
}

// ----- one dense layer: Y(16xN) = act(X(16xK) @ W^T + bias), in-place in X --
// A fragment (16-bit A 16x32): lane row = lane&15;
//   K runs {kt*32 + hi*8 .. +7} and {kt*32 + hi*8 + 16 .. +23}  (hi = lane>>4)
// B fragment (16-bit B 32x16): lane col N; K runs same as A (hi*8 .. +7, hi*8+16 .. +23)
// C/D (f32 16x16): vgpr r -> row r + hi*8, col = lane&15 (+16*nt)
template <int NKT, int NNT, bool RELU>
__device__ __forceinline__ void mlp_layer(_Float16* __restrict__ X,
                                          const _Float16* __restrict__ W,
                                          int kstride, const float* __restrict__ bias,
                                          int lane) {
  const int col = lane & 15;
  const int hi  = lane >> 4;
  v16h A[NKT];
#pragma unroll
  for (int kt = 0; kt < NKT; ++kt) {
    const _Float16* p  = X + col * XSTR + kt * 32 + hi * 8;
    v8h lo = *(const v8h*)p;
    v8h hg = *(const v8h*)(p + 16);
    A[kt] = __builtin_shufflevector(lo, hg, 0,1,2,3,4,5,6,7,8,9,10,11,12,13,14,15);
  }
#pragma unroll
  for (int nt = 0; nt < NNT; ++nt) {
    const int N = nt * 16 + col;
    v8f acc = {};
#pragma unroll
    for (int kt = 0; kt < NKT; ++kt) {
      const _Float16* wp = W + N * kstride + kt * 32 + hi * 8;
      v8h lo = *(const v8h*)wp;
      v8h hg = *(const v8h*)(wp + 16);
      v16h Bf = __builtin_shufflevector(lo, hg, 0,1,2,3,4,5,6,7,8,9,10,11,12,13,14,15);
      acc = __builtin_amdgcn_wmma_f32_16x16x32_f16(false, A[kt], false, Bf,
                                                   (short)0, acc, false, false);
    }
    const float bn = bias[N];
#pragma unroll
    for (int r = 0; r < 8; ++r) {
      float v = acc[r] + bn;
      if (RELU) v = fmaxf(v, 0.0f);
      X[(r + hi * 8) * XSTR + N] = (_Float16)v;
    }
  }
}

__global__ void __launch_bounds__(256) frame_gnn_kernel(
    const float* __restrict__ state,
    const float* __restrict__ upZ1w, const float* __restrict__ uph1w,
    const float* __restrict__ uph1b, const float* __restrict__ upg1w1,
    const float* __restrict__ upg1b1, const float* __restrict__ upg1w2,
    const float* __restrict__ upg1b2, const float* __restrict__ upZ2w,
    const float* __restrict__ upmw1, const float* __restrict__ upmb1,
    const float* __restrict__ upmw2, const float* __restrict__ upmb2,
    const float* __restrict__ upmw3, const float* __restrict__ upmb3,
    const float* __restrict__ dng1w1, const float* __restrict__ dng1b1,
    const float* __restrict__ dng1w2, const float* __restrict__ dng1b2,
    const float* __restrict__ dnZ2w,
    const float* __restrict__ frw1, const float* __restrict__ frb1,
    const float* __restrict__ frw2, const float* __restrict__ frb2,
    const float* __restrict__ frw3, const float* __restrict__ frb3,
    float* __restrict__ out) {
  extern __shared__ __align__(16) char smem[];
  _Float16* W1 = (_Float16*)(smem + OFF_W1);
  _Float16* W2 = (_Float16*)(smem + OFF_W2);
  _Float16* W3 = (_Float16*)(smem + OFF_W3);
  _Float16* F1 = (_Float16*)(smem + OFF_F1);
  _Float16* F2 = (_Float16*)(smem + OFF_F2);
  _Float16* F3 = (_Float16*)(smem + OFF_F3);
  float* BIAS  = (float*)(smem + OFF_B);
  float* SW    = (float*)(smem + OFF_SW);

  const int tid = threadIdx.x, nthr = blockDim.x;
  // ---- convert weights to f16 in LDS (zero-padded), copy biases/smalls ----
  cpw(W1, upmw1, 64, 88, 64, 96, tid, nthr);
  cpw(W2, upmw2, 64, 64, 64, 64, tid, nthr);
  cpw(W3, upmw3, 72, 64, 80, 64, tid, nthr);
  cpw(F1, frw1, 64, 80, 64, 96, tid, nthr);
  cpw(F2, frw2, 64, 64, 64, 64, tid, nthr);
  cpw(F3, frw3, 8, 64, 16, 64, tid, nthr);
  cpb(BIAS + BO_B1, upmb1, 64, 64, tid, nthr);
  cpb(BIAS + BO_B2, upmb2, 64, 64, tid, nthr);
  cpb(BIAS + BO_B3, upmb3, 72, 80, tid, nthr);
  cpb(BIAS + BO_FB1, frb1, 64, 64, tid, nthr);
  cpb(BIAS + BO_FB2, frb2, 64, 64, tid, nthr);
  cpb(BIAS + BO_FB3, frb3, 8, 16, tid, nthr);
  cpf(SW + SW_UPZ1, upZ1w, 32, tid, nthr);
  cpf(SW + SW_UPH1W, uph1w, 104, tid, nthr);
  cpf(SW + SW_UPH1B, uph1b, 8, tid, nthr);
  cpf(SW + SW_UPG1W1, upg1w1, 192, tid, nthr);
  cpf(SW + SW_UPG1B1, upg1b1, 8, tid, nthr);
  cpf(SW + SW_UPG1W2, upg1w2, 8, tid, nthr);
  cpf(SW + SW_UPG1B2, upg1b2, 1, tid, nthr);
  cpf(SW + SW_UPZ2, upZ2w, 200, tid, nthr);
  cpf(SW + SW_DNG1W1, dng1w1, 128, tid, nthr);
  cpf(SW + SW_DNG1B1, dng1b1, 8, tid, nthr);
  cpf(SW + SW_DNG1W2, dng1w2, 8, tid, nthr);
  cpf(SW + SW_DNG1B2, dng1b2, 1, tid, nthr);
  cpf(SW + SW_DNZ2, dnZ2w, 136, tid, nthr);
  __syncthreads();

  const int wave = tid >> 5, lane = tid & 31;
  _Float16* X = (_Float16*)(smem + OFF_X + wave * XWAVEB);
  float* Sf = (float*)X;  // f32 staging view of the same per-wave tile
  const long ebase = (long)blockIdx.x * 128 + wave * 16;

  // ---- coalesced load of 16 state rows into LDS ----
  for (int i = lane; i < 400; i += 32) Sf[i] = state[ebase * 25 + i];

  // persistent per-element state (valid on lanes 0-15)
  float Z0[3][4] = {}, Zm[3][8] = {}, Zm2[3][8] = {}, h0[13] = {};
  float Fn = 1.0f, Fn2 = 1.0f;

  // ================= phase 0 (scalar, lanes 0-15) =================
  if (lane < 16) {
    float sv[25];
#pragma unroll
    for (int j = 0; j < 25; ++j) sv[j] = Sf[lane * 25 + j];
#pragma unroll
    for (int k = 0; k < 4; ++k)
#pragma unroll
      for (int i = 0; i < 3; ++i) Z0[i][k] = sv[4 + k * 3 + i];
#pragma unroll
    for (int j = 0; j < 4; ++j) h0[j] = sv[j];
#pragma unroll
    for (int j = 4; j < 13; ++j) h0[j] = sv[12 + j];
    float hm8[8];
#pragma unroll
    for (int o = 0; o < 8; ++o) {
      float a = SW[SW_UPH1B + o];
#pragma unroll
      for (int j = 0; j < 13; ++j) a += SW[SW_UPH1W + o * 13 + j] * h0[j];
      hm8[o] = fmaxf(a, 0.0f);
    }
    float gs = SW[SW_UPG1B2];
#pragma unroll
    for (int o = 0; o < 8; ++o) {
      float a = SW[SW_UPG1B1 + o];
#pragma unroll
      for (int j = 0; j < 8; ++j) a += SW[SW_UPG1W1 + o * 24 + j] * hm8[j];
      gs += SW[SW_UPG1W2 + o] * fmaxf(a, 0.0f);
    }
    float Zu[3][8];
#pragma unroll
    for (int i = 0; i < 3; ++i)
#pragma unroll
      for (int m = 0; m < 8; ++m) {
        float a = 0.0f;
#pragma unroll
        for (int k = 0; k < 4; ++k) a += Z0[i][k] * SW[SW_UPZ1 + m * 4 + k];
        Zu[i][m] = a;
      }
#pragma unroll
    for (int i = 0; i < 3; ++i)
#pragma unroll
      for (int k = 0; k < 8; ++k) {
        float a = (i == 2) ? gs * SW[SW_UPZ2 + k * 25 + 24] : 0.0f;
#pragma unroll
        for (int m = 0; m < 8; ++m) a += Zu[i][m] * SW[SW_UPZ2 + k * 25 + m];
        Zm[i][k] = a;
      }
    float fs = 0.0f;
    _Float16* Xr = X + lane * XSTR;
#pragma unroll
    for (int j = 0; j < 8; ++j)
#pragma unroll
      for (int k = 0; k < 8; ++k) {
        float m = Zm[0][j] * Zm[0][k] + Zm[1][j] * Zm[1][k] + Zm[2][j] * Zm[2][k];
        fs += m * m;
        Xr[j * 8 + k] = (_Float16)m;
      }
    Fn = sqrtf(fs) + 1.0f;
#pragma unroll
    for (int o = 0; o < 8; ++o) Xr[64 + o] = (_Float16)hm8[o];
#pragma unroll
    for (int c = 72; c < 96; ++c) Xr[c] = (_Float16)0.0f;
  }

  // ================= up_m MLP: 88->64->64->72 (WMMA) =================
  mlp_layer<3, 4, true >(X, W1, 96, BIAS + BO_B1, lane);
  mlp_layer<2, 4, true >(X, W2, 64, BIAS + BO_B2, lane);
  mlp_layer<2, 5, false>(X, W3, 64, BIAS + BO_B3, lane);

  // ================= mid phase (scalar, lanes 0-15) =================
  if (lane < 16) {
    _Float16* Xr = X + lane * XSTR;
    const float invFn = 1.0f / Fn;
    float upZ[3][8] = {};
#pragma unroll
    for (int j = 0; j < 8; ++j)
#pragma unroll
      for (int k = 0; k < 8; ++k) {
        const float m = (float)Xr[j * 8 + k] * invFn;
        upZ[0][k] += Zm[0][j] * m;
        upZ[1][k] += Zm[1][j] * m;
        upZ[2][k] += Zm[2][j] * m;
      }
    float hm2[8];
#pragma unroll
    for (int o = 0; o < 8; ++o) hm2[o] = fmaxf((float)Xr[64 + o] * invFn, 0.0f);
    float gs2 = SW[SW_DNG1B2];
#pragma unroll
    for (int o = 0; o < 8; ++o) {
      float a = SW[SW_DNG1B1 + o];
#pragma unroll
      for (int j = 0; j < 8; ++j) a += SW[SW_DNG1W1 + o * 16 + j] * hm2[j];
      gs2 += SW[SW_DNG1W2 + o] * fmaxf(a, 0.0f);
    }
#pragma unroll
    for (int i = 0; i < 3; ++i)
#pragma unroll
      for (int k = 0; k < 8; ++k) {
        float a = (i == 2) ? gs2 * SW[SW_DNZ2 + k * 17 + 16] : 0.0f;
#pragma unroll
        for (int j = 0; j < 8; ++j) a += upZ[i][j] * SW[SW_DNZ2 + k * 17 + j];
        Zm2[i][k] = a;
      }
    float fs = 0.0f;
#pragma unroll
    for (int j = 0; j < 8; ++j)
#pragma unroll
      for (int k = 0; k < 8; ++k) {
        float m = Zm2[0][j] * Zm2[0][k] + Zm2[1][j] * Zm2[1][k] + Zm2[2][j] * Zm2[2][k];
        fs += m * m;
        Xr[j * 8 + k] = (_Float16)m;
      }
    Fn2 = sqrtf(fs) + 1.0f;
#pragma unroll
    for (int o = 0; o < 8; ++o) Xr[64 + o] = (_Float16)hm2[o];
#pragma unroll
    for (int c = 72; c < 96; ++c) Xr[c] = (_Float16)0.0f;
  }

  // ================= fr MLP: 80->64->64->8 (WMMA) =================
  mlp_layer<3, 4, true >(X, F1, 96, BIAS + BO_FB1, lane);
  mlp_layer<2, 4, true >(X, F2, 64, BIAS + BO_FB2, lane);
  mlp_layer<2, 1, false>(X, F3, 64, BIAS + BO_FB3, lane);

  // ================= final phase (scalar, lanes 0-15) =================
  float* Of = (float*)X;
  if (lane < 16) {
    _Float16* Xr = X + lane * XSTR;
    const float invFn2 = 1.0f / Fn2;
    float fr[8];
#pragma unroll
    for (int o = 0; o < 8; ++o) fr[o] = (float)Xr[o] * invFn2;
    float fx = 0.0f, fy = 0.0f;
#pragma unroll
    for (int o = 0; o < 8; ++o) { fx += Zm2[0][o] * fr[o]; fy += Zm2[1][o] * fr[o]; }
    const float nrm = sqrtf(fx * fx + fy * fy) + 1e-6f;
    const float ux = fx / nrm, uy = fy / nrm;
    float zr[3][4];
#pragma unroll
    for (int k = 0; k < 4; ++k) {
      zr[0][k] =  ux * Z0[0][k] + uy * Z0[1][k];
      zr[1][k] = -uy * Z0[0][k] + ux * Z0[1][k];
      zr[2][k] =  Z0[2][k];
    }
    float* orow = Of + lane * 25;
#pragma unroll
    for (int j = 0; j < 4; ++j) orow[j] = h0[j];
#pragma unroll
    for (int k = 0; k < 4; ++k)
#pragma unroll
      for (int i = 0; i < 3; ++i) orow[4 + k * 3 + i] = zr[i][k];
#pragma unroll
    for (int j = 4; j < 13; ++j) orow[12 + j] = h0[j];
  }
  // ---- coalesced store of 16 output rows ----
  for (int i = lane; i < 400; i += 32) out[ebase * 25 + i] = Of[i];
}

extern "C" void kernel_launch(void* const* d_in, const int* in_sizes, int n_in,
                              void* d_out, int out_size, void* d_ws, size_t ws_size,
                              hipStream_t stream) {
  (void)n_in; (void)d_ws; (void)ws_size; (void)out_size;
  const float* state  = (const float*)d_in[0];
  const float* upZ1w  = (const float*)d_in[1];
  const float* uph1w  = (const float*)d_in[2];
  const float* uph1b  = (const float*)d_in[3];
  const float* upg1w1 = (const float*)d_in[4];
  const float* upg1b1 = (const float*)d_in[5];
  const float* upg1w2 = (const float*)d_in[6];
  const float* upg1b2 = (const float*)d_in[7];
  const float* upZ2w  = (const float*)d_in[8];
  const float* upmw1  = (const float*)d_in[9];
  const float* upmb1  = (const float*)d_in[10];
  const float* upmw2  = (const float*)d_in[11];
  const float* upmb2  = (const float*)d_in[12];
  const float* upmw3  = (const float*)d_in[13];
  const float* upmb3  = (const float*)d_in[14];
  const float* dng1w1 = (const float*)d_in[15];
  const float* dng1b1 = (const float*)d_in[16];
  const float* dng1w2 = (const float*)d_in[17];
  const float* dng1b2 = (const float*)d_in[18];
  const float* dnZ2w  = (const float*)d_in[19];
  const float* frw1   = (const float*)d_in[20];
  const float* frb1   = (const float*)d_in[21];
  const float* frw2   = (const float*)d_in[22];
  const float* frb2   = (const float*)d_in[23];
  const float* frw3   = (const float*)d_in[24];
  const float* frb3   = (const float*)d_in[25];
  // d_in[26..31] = mb_* : dead code in the reference, unused.

  const int n_elems = in_sizes[0] / 25;     // 262144
  const int grid = n_elems / 128;           // 128 elements per 256-thread block
  frame_gnn_kernel<<<grid, 256, SMEM_TOTAL, stream>>>(
      state, upZ1w, uph1w, uph1b, upg1w1, upg1b1, upg1w2, upg1b2, upZ2w,
      upmw1, upmb1, upmw2, upmb2, upmw3, upmb3,
      dng1w1, dng1b1, dng1w2, dng1b2, dnZ2w,
      frw1, frb1, frw2, frb2, frw3, frb3,
      (float*)d_out);
}
